// QLlamaMLP_60361470378702
// MI455X (gfx1250) — matrix operand order repaired and checked by reading, not compiled, not run on hardware
//
#include <hip/hip_runtime.h>
#include <hip/hip_bf16.h>

// ---------------------------------------------------------------------------
// QLlamaMLP for MI455X (gfx1250): bf16 WMMA GEMMs + SwiGLU + per-token int8
// fake quant + down-proj.  Wave32, v_wmma_f32_16x16x32_bf16, async LDS loads
// (ASYNCcnt) with explicit double buffering.
// ---------------------------------------------------------------------------

typedef __bf16 bf16_t;
typedef __attribute__((ext_vector_type(8)))  bf16_t v8bf;
typedef __attribute__((ext_vector_type(16))) bf16_t v16bf;
typedef __attribute__((ext_vector_type(8)))  float  v8f;

#define M_TOK   8192      // B*S
#define D_DIM   4096
#define F_DIM   11008
#define QMAX    127.0f

// ---------------------------------------------------------------------------
// CDNA5 async memory->LDS (tracked by ASYNCcnt, no data VGPRs touched)
// ---------------------------------------------------------------------------
__device__ inline unsigned lds_off_u32(const void* p) {
    // generic LDS address = {SHARED_BASE[63:32], byte_offset[31:0]}
    return (unsigned)(unsigned long long)(size_t)p;
}

__device__ inline void async_load_b128(unsigned lds_off, const void* gaddr) {
    asm volatile("global_load_async_to_lds_b128 %0, %1, off"
                 :: "v"(lds_off), "v"(gaddr) : "memory");
}

__device__ inline void wait_async0() {
    asm volatile("s_wait_asynccnt 0" ::: "memory");
}

// ---------------------------------------------------------------------------
// WMMA fragment helpers (LDS -> VGPR, per ISA 16-bit A/B layouts).
//
// A (16x32, MxK): lanes 0-15 hold row M=lane, K=[0..7]+[16..23];
//                 lanes 16-31 hold row M=lane-16, K=[8..15]+[24..31].
// B (32x16, KxN): lanes 0-15 hold col N=lane, K=[0..7]+[16..23];
//                 lanes 16-31 hold col N=lane-16, K=[8..15]+[24..31].
// LDS tiles row-major, K contiguous (stride 32 bf16 = 64 B):
// each fragment = two 16-byte LDS loads per lane.
// ---------------------------------------------------------------------------
union FragU { v16bf v; v8bf h[2]; };

__device__ inline v16bf load_a_frag(const bf16_t* row, int half) {
    FragU f;
    f.h[0] = *(const v8bf*)(row + 8 * half);
    f.h[1] = *(const v8bf*)(row + 16 + 8 * half);
    return f.v;
}

__device__ inline v16bf load_b_frag(const bf16_t* col_row, int half) {
    FragU f;
    f.h[0] = *(const v8bf*)(col_row + 8 * half);
    f.h[1] = *(const v8bf*)(col_row + 16 + 8 * half);
    return f.v;
}

__device__ inline v8f wmma_bf16(v16bf a, v16bf b, v8f c) {
    return __builtin_amdgcn_wmma_f32_16x16x32_bf16(
        false, a, false, b, (short)0, c, false, false);
}

__device__ inline float silu_f(float g) {
    return g / (1.0f + __expf(-g));
}

// ---------------------------------------------------------------------------
// fp32 -> bf16 conversion (grid-stride)
// ---------------------------------------------------------------------------
__global__ __launch_bounds__(256)
void cvt_f32_to_bf16(const float* __restrict__ in, bf16_t* __restrict__ out, long n) {
    long i = (long)blockIdx.x * blockDim.x + threadIdx.x;
    long stride = (long)gridDim.x * blockDim.x;
    for (; i < n; i += stride) out[i] = (bf16_t)in[i];
}

// ---------------------------------------------------------------------------
// Fused gate/up projection: t[M,F] = silu(x*Wg^T)*(x*Wu^T), bf16 out.
// Block tile 64(M) x 128(N); 8 waves = 2(M) x 4(N); wave tile 32x32.
// Double-buffered LDS stages filled by async-to-LDS loads.
// ---------------------------------------------------------------------------
__global__ __launch_bounds__(256)
void gateup_gemm(const bf16_t* __restrict__ xb,   // [M, K=D]
                 const bf16_t* __restrict__ wgb,  // [F, K=D]
                 const bf16_t* __restrict__ wub,  // [F, K=D]
                 bf16_t* __restrict__ t)          // [M, F]
{
    const int K = D_DIM;
    const int n0 = blockIdx.x * 128;  // over F
    const int m0 = blockIdx.y * 64;   // over M

    __shared__ __align__(16) bf16_t As[2][64][32];    //  8 KB
    __shared__ __align__(16) bf16_t Bg[2][128][32];   // 16 KB
    __shared__ __align__(16) bf16_t Bu[2][128][32];   // 16 KB

    const int tid  = threadIdx.x;
    const int lane = tid & 31;
    const int wave = tid >> 5;
    const int wm   = wave & 1;        // 32-row slab
    const int wn   = wave >> 1;       // 32-col slab
    const int half = lane >> 4;
    const int l16  = lane & 15;

    const int srow = tid >> 2;        // 0..63
    const int scol = (tid & 3) * 8;   // 0,8,16,24

    v8f accg[2][2] = {};
    v8f accu[2][2] = {};

    // stage k-chunk into buffer b
    auto stage = [&](int b, int k0) {
        async_load_b128(lds_off_u32(&As[b][srow][scol]),
                        &xb [(long)(m0 + srow) * K + k0 + scol]);
        async_load_b128(lds_off_u32(&Bg[b][srow][scol]),
                        &wgb[(long)(n0 + srow) * K + k0 + scol]);
        async_load_b128(lds_off_u32(&Bg[b][srow + 64][scol]),
                        &wgb[(long)(n0 + srow + 64) * K + k0 + scol]);
        async_load_b128(lds_off_u32(&Bu[b][srow][scol]),
                        &wub[(long)(n0 + srow) * K + k0 + scol]);
        async_load_b128(lds_off_u32(&Bu[b][srow + 64][scol]),
                        &wub[(long)(n0 + srow + 64) * K + k0 + scol]);
    };

    stage(0, 0);
    wait_async0();
    __syncthreads();

    int buf = 0;
    for (int k0 = 0; k0 < K; k0 += 32) {
        const int nk = k0 + 32;
        if (nk < K) stage(buf ^ 1, nk);   // prefetch next stage (async)

        v16bf bg[2], bu[2], a[2];
        #pragma unroll
        for (int s = 0; s < 2; ++s) {
            bg[s] = load_b_frag(&Bg[buf][wn * 32 + s * 16 + l16][0], half);
            bu[s] = load_b_frag(&Bu[buf][wn * 32 + s * 16 + l16][0], half);
        }
        #pragma unroll
        for (int mt = 0; mt < 2; ++mt)
            a[mt] = load_a_frag(&As[buf][wm * 32 + mt * 16 + l16][0], half);

        #pragma unroll
        for (int mt = 0; mt < 2; ++mt)
            #pragma unroll
            for (int s = 0; s < 2; ++s) {
                accg[mt][s] = wmma_bf16(a[mt], bg[s], accg[mt][s]);
                accu[mt][s] = wmma_bf16(a[mt], bu[s], accu[mt][s]);
            }

        wait_async0();      // next stage resident
        __syncthreads();    // all waves done reading current stage
        buf ^= 1;
    }

    // Epilogue: silu(g)*u -> bf16.  C layout: VGPR r of lane l holds
    // (row = r + 8*(l>>4), col = l&15) of the 16x16 tile.
    #pragma unroll
    for (int mt = 0; mt < 2; ++mt) {
        #pragma unroll
        for (int s = 0; s < 2; ++s) {
            const int col = n0 + wn * 32 + s * 16 + l16;
            #pragma unroll
            for (int r = 0; r < 8; ++r) {
                const int row = m0 + wm * 32 + mt * 16 + r + 8 * half;
                const float g = accg[mt][s][r];
                const float u = accu[mt][s][r];
                t[(long)row * F_DIM + col] = (bf16_t)(silu_f(g) * u);
            }
        }
    }
}

// ---------------------------------------------------------------------------
// Fused per-token amax + int8 fake quant (one workgroup per token row).
// Row staged through LDS: single global read, single global write.
// ---------------------------------------------------------------------------
__global__ __launch_bounds__(256)
void quant_rows(bf16_t* __restrict__ t) {
    __shared__ __align__(16) bf16_t row[F_DIM];   // 22016 B
    __shared__ float red[256];
    const long base = (long)blockIdx.x * F_DIM;

    float m = 0.0f;
    for (int f = threadIdx.x; f < F_DIM; f += 256) {
        const bf16_t v = t[base + f];
        row[f] = v;
        m = fmaxf(m, fabsf((float)v));
    }
    red[threadIdx.x] = m;
    __syncthreads();
    for (int s = 128; s > 0; s >>= 1) {
        if (threadIdx.x < s) red[threadIdx.x] = fmaxf(red[threadIdx.x], red[threadIdx.x + s]);
        __syncthreads();
    }
    const float scale = fmaxf(red[0] / QMAX, 1e-8f);

    for (int f = threadIdx.x; f < F_DIM; f += 256) {
        float q = rintf((float)row[f] / scale);
        q = fminf(fmaxf(q, -QMAX - 1.0f), QMAX);
        t[base + f] = (bf16_t)(q * scale);
    }
}

// ---------------------------------------------------------------------------
// Down projection: out[M,D] = tq[M,F] * Wd[D,F]^T, fp32 out.
// Block tile 64(M) x 128(N); wave tile 32x32; async double buffering.
// ---------------------------------------------------------------------------
__global__ __launch_bounds__(256)
void down_gemm(const bf16_t* __restrict__ tq,   // [M, K=F]
               const bf16_t* __restrict__ wdb,  // [D, K=F]
               float* __restrict__ out)         // [M, D]
{
    const int K = F_DIM;
    const int n0 = blockIdx.x * 128;  // over D
    const int m0 = blockIdx.y * 64;   // over M

    __shared__ __align__(16) bf16_t As[2][64][32];    //  8 KB
    __shared__ __align__(16) bf16_t Bs[2][128][32];   // 16 KB

    const int tid  = threadIdx.x;
    const int lane = tid & 31;
    const int wave = tid >> 5;
    const int wm   = wave & 1;
    const int wn   = wave >> 1;
    const int half = lane >> 4;
    const int l16  = lane & 15;

    const int srow = tid >> 2;
    const int scol = (tid & 3) * 8;

    v8f acc[2][2] = {};

    auto stage = [&](int b, int k0) {
        async_load_b128(lds_off_u32(&As[b][srow][scol]),
                        &tq [(long)(m0 + srow) * K + k0 + scol]);
        async_load_b128(lds_off_u32(&Bs[b][srow][scol]),
                        &wdb[(long)(n0 + srow) * K + k0 + scol]);
        async_load_b128(lds_off_u32(&Bs[b][srow + 64][scol]),
                        &wdb[(long)(n0 + srow + 64) * K + k0 + scol]);
    };

    stage(0, 0);
    wait_async0();
    __syncthreads();

    int buf = 0;
    for (int k0 = 0; k0 < K; k0 += 32) {
        const int nk = k0 + 32;
        if (nk < K) stage(buf ^ 1, nk);

        v16bf b[2], a[2];
        #pragma unroll
        for (int s = 0; s < 2; ++s)
            b[s] = load_b_frag(&Bs[buf][wn * 32 + s * 16 + l16][0], half);
        #pragma unroll
        for (int mt = 0; mt < 2; ++mt)
            a[mt] = load_a_frag(&As[buf][wm * 32 + mt * 16 + l16][0], half);

        #pragma unroll
        for (int mt = 0; mt < 2; ++mt)
            #pragma unroll
            for (int s = 0; s < 2; ++s)
                acc[mt][s] = wmma_bf16(a[mt], b[s], acc[mt][s]);

        wait_async0();
        __syncthreads();
        buf ^= 1;
    }

    #pragma unroll
    for (int mt = 0; mt < 2; ++mt) {
        #pragma unroll
        for (int s = 0; s < 2; ++s) {
            const int col = n0 + wn * 32 + s * 16 + l16;
            #pragma unroll
            for (int r = 0; r < 8; ++r) {
                const int row = m0 + wm * 32 + mt * 16 + r + 8 * half;
                out[(long)row * D_DIM + col] = acc[mt][s][r];
            }
        }
    }
}

// ---------------------------------------------------------------------------
// Launch
// ---------------------------------------------------------------------------
extern "C" void kernel_launch(void* const* d_in, const int* in_sizes, int n_in,
                              void* d_out, int out_size, void* d_ws, size_t ws_size,
                              hipStream_t stream) {
    const float* x  = (const float*)d_in[0];   // [4,2048,4096]
    const float* wg = (const float*)d_in[1];   // [11008,4096]
    const float* wu = (const float*)d_in[2];   // [11008,4096]
    const float* wd = (const float*)d_in[3];   // [4096,11008]
    float* out = (float*)d_out;                // [4,2048,4096]

    const long n_x = (long)M_TOK * D_DIM;
    const long n_w = (long)F_DIM * D_DIM;
    const long n_t = (long)M_TOK * F_DIM;

    char* ws = (char*)d_ws;
    bf16_t* xb  = (bf16_t*)ws;  ws += n_x * sizeof(bf16_t);
    bf16_t* wgb = (bf16_t*)ws;  ws += n_w * sizeof(bf16_t);
    bf16_t* wub = (bf16_t*)ws;  ws += n_w * sizeof(bf16_t);
    bf16_t* wdb = (bf16_t*)ws;  ws += n_w * sizeof(bf16_t);
    bf16_t* t   = (bf16_t*)ws;

    cvt_f32_to_bf16<<<4096, 256, 0, stream>>>(x,  xb,  n_x);
    cvt_f32_to_bf16<<<4096, 256, 0, stream>>>(wg, wgb, n_w);
    cvt_f32_to_bf16<<<4096, 256, 0, stream>>>(wu, wub, n_w);
    cvt_f32_to_bf16<<<4096, 256, 0, stream>>>(wd, wdb, n_w);

    gateup_gemm<<<dim3(F_DIM / 128, M_TOK / 64), 256, 0, stream>>>(xb, wgb, wub, t);
    quant_rows<<<M_TOK, 256, 0, stream>>>(t);
    down_gemm<<<dim3(D_DIM / 128, M_TOK / 64), 256, 0, stream>>>(t, wdb, out);
}
